// MultiHeadAttention3D_15547781612054
// MI455X (gfx1250) — matrix operand order repaired and checked by reading, not compiled, not run on hardware
//
#include <hip/hip_runtime.h>
#include <hip/hip_bf16.h>

typedef __attribute__((ext_vector_type(16))) _Float16 v16h;
typedef __attribute__((ext_vector_type(8)))  _Float16 v8h;
typedef __attribute__((ext_vector_type(8)))  float    v8f;

#define C_DIM  512
#define L_DIM  1024   // h*w
#define T_DIM  8
#define BT_DIM 16     // b*t
#define HEADS  8
#define DHEAD  64
#define LN_EPS 1e-5f

// ---------------- WMMA helpers (CDNA5 gfx1250, wave32) ----------------

__device__ __forceinline__ v8f wmma32(v16h a, v16h b, v8f c) {
  // D(16x16 f32) = A(16x32 f16) * B(32x16 f16) + C
  return __builtin_amdgcn_wmma_f32_16x16x32_f16(
      /*neg_a=*/false, a, /*neg_b=*/false, b,
      /*c_mod=*/(short)0, c, /*reuse_a=*/false, /*reuse_b=*/false);
}

// A fragment: 16x32 f16, row-major source with leading dim `ld`.
// ISA 7.12.2: lane m = lane&15; lanes<16 hold K={0..7,16..23}, lanes>=16 hold K={8..15,24..31}.
__device__ __forceinline__ v16h load_a_frag(const _Float16* A, int ld) {
  int lane = threadIdx.x & 31;
  int m  = lane & 15;
  int kb = (lane >> 4) << 3;           // 0 or 8
  const _Float16* p = A + (size_t)m * ld + kb;
  v8h lo = *(const v8h*)(p);           // K = kb .. kb+7
  v8h hi = *(const v8h*)(p + 16);      // K = kb+16 .. kb+23
  v16h f;
#pragma unroll
  for (int i = 0; i < 8; ++i) { f[i] = lo[i]; f[i + 8] = hi[i]; }
  return f;
}

// B fragment from B^T (N x K row-major, leading dim `ld`).
// ISA: lane n = lane&15; lanes<16 hold K={0..7,16..23}, lanes>=16 hold K={8..15,24..31}.
__device__ __forceinline__ v16h load_bt_frag(const _Float16* BT, int ld) {
  int lane = threadIdx.x & 31;
  int n  = lane & 15;
  int kb = (lane >> 4) << 3;           // 0 or 8
  const _Float16* p = BT + (size_t)n * ld + kb;
  v8h lo = *(const v8h*)(p);
  v8h hi = *(const v8h*)(p + 16);
  v16h f;
#pragma unroll
  for (int i = 0; i < 8; ++i) { f[i] = lo[i]; f[i + 8] = hi[i]; }
  return f;
}

// ---------------- 1) LayerNorm + f16 tokenization ----------------
// x: (b, c, t, h, w) fp32  ->  n: [bt][l][c] f16
__global__ __launch_bounds__(256) void ln_kernel(
    const float* __restrict__ x, const float* __restrict__ g,
    const float* __restrict__ be, _Float16* __restrict__ nout)
{
  int tok = blockIdx.x;                 // 0 .. BT*L-1
  int bt = tok >> 10, l = tok & 1023;
  int b = bt >> 3, t = bt & 7;
  const float* xb = x + ((size_t)b * C_DIM * T_DIM + t) * L_DIM + l;  // stride T*L per channel
  int tid = threadIdx.x;                // 256 threads, 2 channels each
  float v0 = xb[(size_t)tid * T_DIM * L_DIM];
  float v1 = xb[(size_t)(tid + 256) * T_DIM * L_DIM];
  float s = v0 + v1, ss = v0 * v0 + v1 * v1;
#pragma unroll
  for (int off = 16; off; off >>= 1) {
    s  += __shfl_xor(s,  off, 32);
    ss += __shfl_xor(ss, off, 32);
  }
  __shared__ float sh[16];
  int w = tid >> 5;
  if ((tid & 31) == 0) { sh[w] = s; sh[8 + w] = ss; }
  __syncthreads();
  float S = 0.f, SS = 0.f;
#pragma unroll
  for (int i = 0; i < 8; ++i) { S += sh[i]; SS += sh[8 + i]; }
  float mu   = S * (1.0f / C_DIM);
  float var  = SS * (1.0f / C_DIM) - mu * mu;
  float rstd = rsqrtf(var + LN_EPS);
  _Float16* o = nout + (size_t)tok * C_DIM;
  o[tid]       = (_Float16)((v0 - mu) * rstd * g[tid] + be[tid]);
  o[tid + 256] = (_Float16)((v1 - mu) * rstd * g[tid + 256] + be[tid + 256]);
}

// ---------------- 2) Weight convert + transpose: WT[n][k] = (f16) W[k][n] ----------------
__global__ __launch_bounds__(256) void wcvt(
    const float* __restrict__ W, _Float16* __restrict__ WT)
{
  __shared__ float tile[16][17];
  int tx = threadIdx.x & 15, ty = threadIdx.x >> 4;
  int k0 = blockIdx.y * 16, n0 = blockIdx.x * 16;
  tile[ty][tx] = W[(size_t)(k0 + ty) * C_DIM + n0 + tx];
  __syncthreads();
  WT[(size_t)(n0 + ty) * C_DIM + k0 + tx] = (_Float16)tile[tx][ty];
}

// ---------------- 3) QKV projection GEMM (32x32 macro-tile per wave) ----------------
// out = scale * (A @ W + bias) ; A=[bt][L][C] f16, WT = W^T f16.
// transpose_out==0 : out[bt][l][c]      (q, k)
// transpose_out==1 : out[bt][c][l]      (v, pre-transposed for attn*v)
__global__ __launch_bounds__(256) void qkv_gemm(
    const _Float16* __restrict__ A, const _Float16* __restrict__ WT,
    const float* __restrict__ bias, _Float16* __restrict__ out,
    float scale, int transpose_out)
{
  const int tilesN = C_DIM / 32, tilesM = L_DIM / 32;          // 16, 32
  int wave = blockIdx.x * (blockDim.x >> 5) + (threadIdx.x >> 5);
  int bt  = wave / (tilesM * tilesN);
  int rem = wave % (tilesM * tilesN);
  int tm = rem / tilesN, tn = rem % tilesN;
  const _Float16* A0 = A  + (size_t)bt * L_DIM * C_DIM + (size_t)tm * 32 * C_DIM;
  const _Float16* A1 = A0 + (size_t)16 * C_DIM;
  const _Float16* B0 = WT + (size_t)tn * 32 * C_DIM;
  const _Float16* B1 = B0 + (size_t)16 * C_DIM;
  v8f acc00 = {}, acc01 = {}, acc10 = {}, acc11 = {};
#pragma unroll
  for (int k = 0; k < C_DIM; k += 32) {
    __builtin_prefetch(B0 + k + 128, 0, 3);        // global_prefetch_b8 (keep near)
    __builtin_prefetch(B1 + k + 128, 0, 3);
    v16h a0 = load_a_frag(A0 + k, C_DIM);
    v16h a1 = load_a_frag(A1 + k, C_DIM);
    v16h b0 = load_bt_frag(B0 + k, C_DIM);
    v16h b1 = load_bt_frag(B1 + k, C_DIM);
    acc00 = wmma32(a0, b0, acc00);
    acc01 = wmma32(a0, b1, acc01);
    acc10 = wmma32(a1, b0, acc10);
    acc11 = wmma32(a1, b1, acc11);
  }
  int lane = threadIdx.x & 31;
  int n  = lane & 15;
  v8f accs[2][2] = {{acc00, acc01}, {acc10, acc11}};
#pragma unroll
  for (int r = 0; r < 2; ++r) {
#pragma unroll
    for (int c = 0; c < 2; ++c) {
      int cn = tn * 32 + c * 16 + n;
      float bi = bias[cn];
#pragma unroll
      for (int i = 0; i < 8; ++i) {
        int m = tm * 32 + r * 16 + i + ((lane >> 4) << 3);
        float val = (accs[r][c][i] + bi) * scale;
        if (!transpose_out)
          out[(size_t)bt * L_DIM * C_DIM + (size_t)m * C_DIM + cn] = (_Float16)val;
        else
          out[(size_t)bt * C_DIM * L_DIM + (size_t)cn * L_DIM + m] = (_Float16)val;
      }
    }
  }
}

// ---------------- 4) Flash attention: softmax(q k^T) v ----------------
// q: [bt][l][c] (pre-scaled by 1/sqrt(d)), k: [bt][l][c], vt: [bt][c][l]
// o: [bt][l][c] f16. Block = 4 waves x 32 rows = 128 query rows per block.
__global__ __launch_bounds__(128) void attn_kernel(
    const _Float16* __restrict__ q, const _Float16* __restrict__ k,
    const _Float16* __restrict__ vt, _Float16* __restrict__ o)
{
  __shared__ __align__(16) _Float16 plds[4 * 32 * 32];
  int unit = blockIdx.x;
  int mb = unit & 7;   unit >>= 3;      // row block (128 rows)
  int hd = unit & 7;   unit >>= 3;      // head
  int bt = unit;
  int wave = threadIdx.x >> 5;
  int lane = threadIdx.x & 31;
  int r0 = mb * 128 + wave * 32;        // 32 rows per wave

  const _Float16* qb = q  + (size_t)bt * L_DIM * C_DIM + (size_t)r0 * C_DIM + hd * DHEAD;
  const _Float16* kb = k  + (size_t)bt * L_DIM * C_DIM + hd * DHEAD;
  const _Float16* vb = vt + (size_t)bt * C_DIM * L_DIM + (size_t)hd * DHEAD * L_DIM;

  // q fragments for 2 row tiles x K=64 (resident across the whole L sweep)
  v16h aq[2][2];
#pragma unroll
  for (int r = 0; r < 2; ++r) {
    aq[r][0] = load_a_frag(qb + (size_t)r * 16 * C_DIM,      C_DIM);
    aq[r][1] = load_a_frag(qb + (size_t)r * 16 * C_DIM + 32, C_DIM);
  }

  v8f O[2][4];
  float mrow[2][8], lrow[2][8];
#pragma unroll
  for (int r = 0; r < 2; ++r) {
#pragma unroll
    for (int nt = 0; nt < 4; ++nt) O[r][nt] = (v8f){};
#pragma unroll
    for (int i = 0; i < 8; ++i) { mrow[r][i] = -1e30f; lrow[r][i] = 0.f; }
  }

  _Float16* pl = plds + wave * 32 * 32;

  for (int l0 = 0; l0 < L_DIM; l0 += 32) {
    // ---- key fragments for this 32-column chunk (shared by both row tiles)
    v16h bk00 = load_bt_frag(kb + (size_t)l0 * C_DIM,             C_DIM);
    v16h bk01 = load_bt_frag(kb + (size_t)l0 * C_DIM + 32,        C_DIM);
    v16h bk10 = load_bt_frag(kb + (size_t)(l0 + 16) * C_DIM,      C_DIM);
    v16h bk11 = load_bt_frag(kb + (size_t)(l0 + 16) * C_DIM + 32, C_DIM);

#pragma unroll
    for (int r = 0; r < 2; ++r) {
      // ---- S tiles (16x32): K = d = 64 -> 2 WMMAs per 16-col tile
      v8f S0 = {}, S1 = {};
      S0 = wmma32(aq[r][0], bk00, S0);
      S0 = wmma32(aq[r][1], bk01, S0);
      S1 = wmma32(aq[r][0], bk10, S1);
      S1 = wmma32(aq[r][1], bk11, S1);

      // ---- online softmax; a row lives across 16 lanes of a half-wave
#pragma unroll
      for (int i = 0; i < 8; ++i) {
        float tmax = fmaxf(S0[i], S1[i]);
#pragma unroll
        for (int off = 8; off; off >>= 1) tmax = fmaxf(tmax, __shfl_xor(tmax, off, 32));
        float nm   = fmaxf(mrow[r][i], tmax);
        float corr = __expf(mrow[r][i] - nm);
        float p0   = __expf(S0[i] - nm);
        float p1   = __expf(S1[i] - nm);
        float ts   = p0 + p1;
#pragma unroll
        for (int off = 8; off; off >>= 1) ts += __shfl_xor(ts, off, 32);
        lrow[r][i] = lrow[r][i] * corr + ts;
        mrow[r][i] = nm;
        O[r][0][i] *= corr; O[r][1][i] *= corr;
        O[r][2][i] *= corr; O[r][3][i] *= corr;
        // stash P tile (C-layout -> row-major 32x32 in LDS)
        int m = r * 16 + i + ((lane >> 4) << 3);
        int n = lane & 15;
        pl[m * 32 + n]      = (_Float16)p0;
        pl[m * 32 + 16 + n] = (_Float16)p1;
      }
    }

    // ---- value fragments for this chunk (shared by both row tiles)
    v16h bv0 = load_bt_frag(vb + (size_t)0 * 16 * L_DIM + l0, L_DIM);
    v16h bv1 = load_bt_frag(vb + (size_t)1 * 16 * L_DIM + l0, L_DIM);
    v16h bv2 = load_bt_frag(vb + (size_t)2 * 16 * L_DIM + l0, L_DIM);
    v16h bv3 = load_bt_frag(vb + (size_t)3 * 16 * L_DIM + l0, L_DIM);

    // ---- O += P(16x32) @ V(32x64): reload P as A fragments from LDS
#pragma unroll
    for (int r = 0; r < 2; ++r) {
      v16h ap = load_a_frag(pl + (size_t)r * 16 * 32, 32);
      O[r][0] = wmma32(ap, bv0, O[r][0]);
      O[r][1] = wmma32(ap, bv1, O[r][1]);
      O[r][2] = wmma32(ap, bv2, O[r][2]);
      O[r][3] = wmma32(ap, bv3, O[r][3]);
    }
  }

  // ---- epilogue: normalize and write [bt][l][head*64 + dd]
  _Float16* ob = o + (size_t)bt * L_DIM * C_DIM + (size_t)r0 * C_DIM + hd * DHEAD;
  int n = lane & 15;
#pragma unroll
  for (int r = 0; r < 2; ++r) {
#pragma unroll
    for (int i = 0; i < 8; ++i) {
      int m = r * 16 + i + ((lane >> 4) << 3);
      float inv = 1.0f / lrow[r][i];
      ob[(size_t)m * C_DIM +  0 + n] = (_Float16)(O[r][0][i] * inv);
      ob[(size_t)m * C_DIM + 16 + n] = (_Float16)(O[r][1][i] * inv);
      ob[(size_t)m * C_DIM + 32 + n] = (_Float16)(O[r][2][i] * inv);
      ob[(size_t)m * C_DIM + 48 + n] = (_Float16)(O[r][3][i] * inv);
    }
  }
}

// ---------------- 5) Output projection + residual (32x32 macro-tile per wave) ----------------
// out(fp32, x-layout) = x + A @ Wp + bp ; A = attn tokens f16 [bt][l][c]
__global__ __launch_bounds__(256) void proj_kernel(
    const _Float16* __restrict__ A, const _Float16* __restrict__ WT,
    const float* __restrict__ bias, const float* __restrict__ x,
    float* __restrict__ out)
{
  const int tilesN = C_DIM / 32, tilesM = L_DIM / 32;
  int wave = blockIdx.x * (blockDim.x >> 5) + (threadIdx.x >> 5);
  int bt  = wave / (tilesM * tilesN);
  int rem = wave % (tilesM * tilesN);
  int tm = rem / tilesN, tn = rem % tilesN;
  const _Float16* A0 = A  + (size_t)bt * L_DIM * C_DIM + (size_t)tm * 32 * C_DIM;
  const _Float16* A1 = A0 + (size_t)16 * C_DIM;
  const _Float16* B0 = WT + (size_t)tn * 32 * C_DIM;
  const _Float16* B1 = B0 + (size_t)16 * C_DIM;
  v8f acc00 = {}, acc01 = {}, acc10 = {}, acc11 = {};
#pragma unroll
  for (int kk = 0; kk < C_DIM; kk += 32) {
    __builtin_prefetch(B0 + kk + 128, 0, 3);
    __builtin_prefetch(B1 + kk + 128, 0, 3);
    v16h a0 = load_a_frag(A0 + kk, C_DIM);
    v16h a1 = load_a_frag(A1 + kk, C_DIM);
    v16h b0 = load_bt_frag(B0 + kk, C_DIM);
    v16h b1 = load_bt_frag(B1 + kk, C_DIM);
    acc00 = wmma32(a0, b0, acc00);
    acc01 = wmma32(a0, b1, acc01);
    acc10 = wmma32(a1, b0, acc10);
    acc11 = wmma32(a1, b1, acc11);
  }
  int lane = threadIdx.x & 31;
  int n  = lane & 15;
  int b = bt >> 3, t = bt & 7;
  v8f accs[2][2] = {{acc00, acc01}, {acc10, acc11}};
#pragma unroll
  for (int r = 0; r < 2; ++r) {
#pragma unroll
    for (int c = 0; c < 2; ++c) {
      int cn = tn * 32 + c * 16 + n;
      float bi = bias[cn];
#pragma unroll
      for (int i = 0; i < 8; ++i) {
        int l = tm * 32 + r * 16 + i + ((lane >> 4) << 3);
        size_t off = (((size_t)b * C_DIM + cn) * T_DIM + t) * L_DIM + l;
        out[off] = x[off] + accs[r][c][i] + bi;
      }
    }
  }
}

// ---------------- launch ----------------
extern "C" void kernel_launch(void* const* d_in, const int* in_sizes, int n_in,
                              void* d_out, int out_size, void* d_ws, size_t ws_size,
                              hipStream_t stream) {
  const float* x    = (const float*)d_in[0];
  const float* ln_g = (const float*)d_in[1];
  const float* ln_b = (const float*)d_in[2];
  const float* Wq   = (const float*)d_in[3];
  const float* bq   = (const float*)d_in[4];
  const float* Wk   = (const float*)d_in[5];
  const float* bk   = (const float*)d_in[6];
  const float* Wv   = (const float*)d_in[7];
  const float* bv   = (const float*)d_in[8];
  const float* Wp   = (const float*)d_in[9];
  const float* bp   = (const float*)d_in[10];
  float* out = (float*)d_out;

  // workspace carve-up (all 256B-aligned)
  const size_t TOK_BYTES = (size_t)BT_DIM * L_DIM * C_DIM * sizeof(_Float16); // 16 MB
  const size_t W_BYTES   = (size_t)C_DIM * C_DIM * sizeof(_Float16);          // 512 KB
  char* ws = (char*)d_ws;
  _Float16* nbuf = (_Float16*)(ws);                          ws += TOK_BYTES;
  _Float16* qbuf = (_Float16*)(ws);                          ws += TOK_BYTES;
  _Float16* kbuf = (_Float16*)(ws);                          ws += TOK_BYTES;
  _Float16* vtbf = (_Float16*)(ws);                          ws += TOK_BYTES;
  _Float16* abuf = (_Float16*)(ws);                          ws += TOK_BYTES;
  _Float16* wtq  = (_Float16*)(ws);                          ws += W_BYTES;
  _Float16* wtk  = (_Float16*)(ws);                          ws += W_BYTES;
  _Float16* wtv  = (_Float16*)(ws);                          ws += W_BYTES;
  _Float16* wtp  = (_Float16*)(ws);                          ws += W_BYTES;

  // 1) LayerNorm -> f16 tokens
  ln_kernel<<<BT_DIM * L_DIM, 256, 0, stream>>>(x, ln_g, ln_b, nbuf);

  // 2) weight convert+transpose
  dim3 wg(C_DIM / 16, C_DIM / 16);
  wcvt<<<wg, 256, 0, stream>>>(Wq, wtq);
  wcvt<<<wg, 256, 0, stream>>>(Wk, wtk);
  wcvt<<<wg, 256, 0, stream>>>(Wv, wtv);
  wcvt<<<wg, 256, 0, stream>>>(Wp, wtp);

  // 3) QKV projections (32x32 macro-tile per wave, 8 waves per block)
  int gemm_blocks = BT_DIM * (L_DIM / 32) * (C_DIM / 32) / 8;   // 1024
  qkv_gemm<<<gemm_blocks, 256, 0, stream>>>(nbuf, wtq, bq, qbuf, 0.125f, 0); // q * 1/sqrt(64)
  qkv_gemm<<<gemm_blocks, 256, 0, stream>>>(nbuf, wtk, bk, kbuf, 1.0f,   0);
  qkv_gemm<<<gemm_blocks, 256, 0, stream>>>(nbuf, wtv, bv, vtbf, 1.0f,   1); // v transposed

  // 4) attention (4 waves x 32 rows = 128 rows per block)
  attn_kernel<<<BT_DIM * HEADS * (L_DIM / 128), 128, 0, stream>>>(qbuf, kbuf, vtbf, abuf);

  // 5) projection + residual back into (b,c,t,h,w)
  proj_kernel<<<gemm_blocks, 256, 0, stream>>>(abuf, wtp, bp, x, out);
}
